// News_embedding_44083544326658
// MI455X (gfx1250) — compile-verified, matrix operand order checked by reading, not run on hardware
//
#include <hip/hip_runtime.h>
#include <hip/hip_bf16.h>

typedef __attribute__((ext_vector_type(16))) _Float16 v16h;
typedef __attribute__((ext_vector_type(8)))  _Float16 v8h;
typedef __attribute__((ext_vector_type(4)))  _Float16 v4h;
typedef __attribute__((ext_vector_type(8)))  float    v8f;
typedef __attribute__((ext_vector_type(4)))  float    v4f;

#define D_DOC 768
#define D_E   100
#define E_NUM 100
#define D_H   128

#define PWA_ELEMS (4 * 2 * 128 * 16)     // entity weight, fragment-packed
#define PWC_ELEMS (24 * 2 * 256 * 16)    // ctx weights (a1||f1), fragment-packed

#define SHUF16(lo, hi) __builtin_shufflevector((lo), (hi), 0, 1, 2, 3, 4, 5, 6, 7, \
                                               8, 9, 10, 11, 12, 13, 14, 15)

// ---------------------------------------------------------------------------
// Kernel 0: pack weights into per-lane WMMA B-fragment order (f16, L2-resident).
//   pwa[((c*2+hh)*128 + n)*16 + j] = W_a1[k, n]   (k>=100 -> 0),  k = c*32+hh*8+j+(j>=8?8:0)
//   pwc[((c*2+hh)*256 + n)*16 + j] = (n<128 ? W_a1 : W_f1)[100 + k, n&127]
// ---------------------------------------------------------------------------
__global__ __launch_bounds__(256) void pack_weights_kernel(
    const float* __restrict__ w_a1, const float* __restrict__ w_f1,
    _Float16* __restrict__ pwa, _Float16* __restrict__ pwc)
{
    int idx = blockIdx.x * 256 + threadIdx.x;
    if (idx < PWA_ELEMS) {
        int j = idx & 15, rest = idx >> 4;
        int n = rest & 127; rest >>= 7;
        int hh = rest & 1, c = rest >> 1;
        int k = c * 32 + hh * 8 + j + ((j >= 8) ? 8 : 0);
        pwa[idx] = (k < D_E) ? (_Float16)w_a1[k * D_H + n] : (_Float16)0.f;
    } else {
        int id2 = idx - PWA_ELEMS;
        if (id2 < PWC_ELEMS) {
            int j = id2 & 15, rest = id2 >> 4;
            int n = rest & 255; rest >>= 8;
            int hh = rest & 1, c = rest >> 1;
            int k = D_E + c * 32 + hh * 8 + j + ((j >= 8) ? 8 : 0);
            const float* w = (n < D_H) ? w_a1 : w_f1;
            pwc[id2] = (_Float16)w[k * D_H + (n & 127)];
        }
    }
}

// ---------------------------------------------------------------------------
// Kernel 1: fused ctx projections (16 batch rows/block, N=256, K=768).
//   ctx_a = ctx @ W_a1[100:868] + b_a1 ; ctx_f = ctx @ W_f1[100:868] + b_f1
// ---------------------------------------------------------------------------
__global__ __launch_bounds__(256) void ctx_proj_kernel(
    const float* __restrict__ ctx,
    const float* __restrict__ b_a1, const float* __restrict__ b_f1,
    const _Float16* __restrict__ pwc,
    float* __restrict__ ctxa, float* __restrict__ ctxf)
{
    __shared__ alignas(16) _Float16 sctx[16 * 776];   // stride 776 f16 = 1552B (16B-mult, bank-skewed)
    const int b0  = blockIdx.x * 16;
    const int tid = threadIdx.x;

    // stage ctx tile (float4 loads -> v4h stores)
    for (int i = tid; i < 16 * (D_DOC / 4); i += 256) {
        int r = i / (D_DOC / 4), c4 = (i - r * (D_DOC / 4)) * 4;
        v4f v = *(const v4f*)(ctx + (size_t)(b0 + r) * D_DOC + c4);
        v4h h; h[0] = (_Float16)v[0]; h[1] = (_Float16)v[1];
        h[2] = (_Float16)v[2]; h[3] = (_Float16)v[3];
        *(v4h*)(sctx + r * 776 + c4) = h;
    }
    __syncthreads();

    const int wave  = tid >> 5, lane = tid & 31;
    const int lrow  = lane & 15;
    const int hh    = lane >> 4;
    const int khalf = hh * 8;
    const int col0  = wave * 32 + lrow;        // global col in [0,256)
    const int col1  = wave * 32 + 16 + lrow;

    // per-thread (wave-uniform) bias/dst resolution: no hot-loop divergence
    const float bias0 = (col0 < D_H) ? b_a1[col0] : b_f1[col0 - D_H];
    const float bias1 = (col1 < D_H) ? b_a1[col1] : b_f1[col1 - D_H];
    float* dst0 = (col0 < D_H) ? (ctxa + col0) : (ctxf + (col0 - D_H));
    float* dst1 = (col1 < D_H) ? (ctxa + col1) : (ctxf + (col1 - D_H));

    v8f acc0 = {}; v8f acc1 = {};
    for (int c = 0; c < 24; ++c) {
        const int kb = c * 32;
        const v8h* ap = (const v8h*)(sctx + lrow * 776 + kb + khalf);
        v16h afrag = SHUF16(ap[0], ap[2]);          // k: khalf+0..7, khalf+16..23
        const v8h* bp0 = (const v8h*)pwc + (size_t)((c * 2 + hh) * 256 + col0) * 2;
        const v8h* bp1 = (const v8h*)pwc + (size_t)((c * 2 + hh) * 256 + col1) * 2;
        v16h bf0 = SHUF16(bp0[0], bp0[1]);
        v16h bf1 = SHUF16(bp1[0], bp1[1]);
        acc0 = __builtin_amdgcn_wmma_f32_16x16x32_f16(false, afrag, false, bf0,
                                                      (short)0, acc0, false, false);
        acc1 = __builtin_amdgcn_wmma_f32_16x16x32_f16(false, afrag, false, bf1,
                                                      (short)0, acc1, false, false);
    }

    #pragma unroll
    for (int i = 0; i < 8; ++i) {
        size_t gb = (size_t)(b0 + i + hh * 8);
        dst0[gb * D_H] = acc0[i] + bias0;
        dst1[gb * D_H] = acc1[i] + bias1;
    }
}

// ---------------------------------------------------------------------------
// Kernel 2: one block per batch row b.
// ---------------------------------------------------------------------------
__global__ __launch_bounds__(256) void news_main_kernel(
    const float* __restrict__ kgat,
    const int*   __restrict__ idx_ent,
    const int*   __restrict__ idx_title,
    const int*   __restrict__ idx_type,
    const float* __restrict__ tab_ent,
    const float* __restrict__ tab_title,
    const float* __restrict__ tab_type,
    const _Float16* __restrict__ pwa,
    const float* __restrict__ w_a2, const float* __restrict__ b_a2,
    const float* __restrict__ w_f1, const float* __restrict__ w_f2,
    const float* __restrict__ b_f2,
    const float* __restrict__ ctxa, const float* __restrict__ ctxf,
    float* __restrict__ out_news, float* __restrict__ out_soft)
{
    __shared__ alignas(16) _Float16 sent[112 * 136]; // stride 136 f16 = 272B (16B-mult, skewed)
    __shared__ alignas(16) _Float16 sh[100 * 136];
    __shared__ float satt[128];
    __shared__ float sred[256];
    __shared__ float sagg[104];
    __shared__ float sz[128];
    __shared__ float sw2[128];

    const int b   = blockIdx.x;
    const int tid = threadIdx.x;

    // ---- build ent in LDS (f16): float4 gather-sum over d, zero padding ----
    for (int i = tid; i < E_NUM * (D_E / 4); i += 256) {
        int e = i / (D_E / 4), dq = (i - e * (D_E / 4)) * 4;
        size_t be = (size_t)b * E_NUM + e;
        const float* p0 = kgat + be * D_E + dq;
        const float* p1 = tab_ent   + (size_t)idx_ent  [be] * D_E + dq;
        const float* p2 = tab_title + (size_t)idx_title[be] * D_E + dq;
        const float* p3 = tab_type  + (size_t)idx_type [be] * D_E + dq;
        v4f a = *(const v4f*)p0, b1 = *(const v4f*)p1,
            c = *(const v4f*)p2, d  = *(const v4f*)p3;
        v4h h;
        #pragma unroll
        for (int j = 0; j < 4; ++j) h[j] = (_Float16)(a[j] + b1[j] + c[j] + d[j]);
        *(v4h*)(sent + e * 136 + dq) = h;
    }
    // zero pad: d in [100,128) for all 112 rows; rows [100,112) for d in [0,100)
    for (int i = tid; i < 112 * 28; i += 256) {
        int e = i / 28, d = 100 + (i - e * 28);
        sent[e * 136 + d] = (_Float16)0.f;
    }
    for (int i = tid; i < 12 * 100; i += 256) {
        int e = 100 + i / 100, d = i - (e - 100) * 100;
        sent[e * 136 + d] = (_Float16)0.f;
    }
    __syncthreads();

    // ---- WMMA: h = ent @ W_a1[0:100,:]  (M=112, N=128, K=128) ----
    const int wave  = tid >> 5, lane = tid & 31;
    const int lrow  = lane & 15;
    const int hh    = lane >> 4;
    const int khalf = hh * 8;
    const int col   = wave * 16 + lrow;     // 0..127

    v8f acc[7];
    #pragma unroll
    for (int t = 0; t < 7; ++t) acc[t] = (v8f){};

    #pragma unroll
    for (int c = 0; c < 4; ++c) {
        const int kb = c * 32;
        const v8h* bp = (const v8h*)pwa + (size_t)((c * 2 + hh) * 128 + col) * 2;
        v16h bfrag = SHUF16(bp[0], bp[1]);
        #pragma unroll
        for (int t = 0; t < 7; ++t) {
            const v8h* ap = (const v8h*)(sent + (t * 16 + lrow) * 136 + kb + khalf);
            v16h afrag = SHUF16(ap[0], ap[2]);
            acc[t] = __builtin_amdgcn_wmma_f32_16x16x32_f16(false, afrag, false, bfrag,
                                                            (short)0, acc[t], false, false);
        }
    }

    // ---- epilogue: + ctx_a, ReLU, store h (f16); stage W_a2 to LDS ----
    const float ca = ctxa[(size_t)b * D_H + col];
    #pragma unroll
    for (int t = 0; t < 7; ++t) {
        #pragma unroll
        for (int i = 0; i < 8; ++i) {
            int m = t * 16 + i + hh * 8;
            if (m < E_NUM) {
                float v = acc[t][i] + ca;
                sh[m * 136 + col] = (_Float16)(v > 0.f ? v : 0.f);
            }
        }
    }
    if (tid < D_H) sw2[tid] = w_a2[tid];
    __syncthreads();

    // ---- att[e] = relu(h[e] . W_a2 + b_a2) ----
    if (tid < E_NUM) {
        const v8h* hrow = (const v8h*)(sh + tid * 136);
        float s = 0.f;
        #pragma unroll
        for (int c8 = 0; c8 < 16; ++c8) {
            v8h hv = hrow[c8];
            #pragma unroll
            for (int j = 0; j < 8; ++j) s += (float)hv[j] * sw2[c8 * 8 + j];
        }
        s += b_a2[0];
        satt[tid] = s > 0.f ? s : 0.f;
    }
    __syncthreads();

    // ---- softmax over entities (masked to 100) ----
    sred[tid] = (tid < E_NUM) ? satt[tid] : -1e30f;
    __syncthreads();
    for (int s2 = 128; s2 > 0; s2 >>= 1) {
        if (tid < s2) sred[tid] = fmaxf(sred[tid], sred[tid + s2]);
        __syncthreads();
    }
    const float mx = sred[0];
    __syncthreads();
    const float ex = (tid < E_NUM) ? __expf(satt[tid] - mx) : 0.f;
    sred[tid] = ex;
    __syncthreads();
    for (int s2 = 128; s2 > 0; s2 >>= 1) {
        if (tid < s2) sred[tid] += sred[tid + s2];
        __syncthreads();
    }
    const float inv = 1.f / sred[0];
    __syncthreads();
    const float sm = ex * inv;
    if (tid < E_NUM) {
        satt[tid] = sm;
        out_soft[(size_t)b * E_NUM + tid] = sm;
    }
    __syncthreads();

    // ---- agg[d] = sum_e soft[e] * ent[e,d] ----
    if (tid < D_E) {
        float a = 0.f;
        for (int e = 0; e < E_NUM; ++e) a += satt[e] * (float)sent[e * 136 + tid];
        sagg[tid] = a;
    }
    __syncthreads();

    // ---- z = relu(agg @ W_f1[0:100,:] + ctx_f[b]) ----
    if (tid < D_H) {
        float z = ctxf[(size_t)b * D_H + tid];
        for (int d = 0; d < D_E; ++d) z += sagg[d] * w_f1[d * D_H + tid];
        sz[tid] = z > 0.f ? z : 0.f;
    }
    __syncthreads();

    // ---- news = tanh(z @ W_f2 + b_f2) ----
    if (tid < D_E) {
        float o = b_f2[tid];
        for (int j = 0; j < D_H; ++j) o += sz[j] * w_f2[j * D_E + tid];
        out_news[(size_t)b * D_E + tid] = tanhf(o);
    }
}

extern "C" void kernel_launch(void* const* d_in, const int* in_sizes, int n_in,
                              void* d_out, int out_size, void* d_ws, size_t ws_size,
                              hipStream_t stream) {
    const float* kgat      = (const float*)d_in[0];
    const float* ctx       = (const float*)d_in[1];
    const int*   idx_ent   = (const int*)  d_in[2];
    const int*   idx_title = (const int*)  d_in[3];
    const int*   idx_type  = (const int*)  d_in[4];
    const float* tab_ent   = (const float*)d_in[5];
    const float* tab_title = (const float*)d_in[6];
    const float* tab_type  = (const float*)d_in[7];
    const float* w_a1      = (const float*)d_in[8];
    const float* b_a1      = (const float*)d_in[9];
    const float* w_a2      = (const float*)d_in[10];
    const float* b_a2      = (const float*)d_in[11];
    const float* w_f1      = (const float*)d_in[12];
    const float* b_f1      = (const float*)d_in[13];
    const float* w_f2      = (const float*)d_in[14];
    const float* b_f2      = (const float*)d_in[15];

    const int B = in_sizes[1] / D_DOC;   // 8192

    float* out_news = (float*)d_out;                     // [B,100]
    float* out_soft = out_news + (size_t)B * E_NUM;      // [B,100,1]

    // workspace layout
    float*    ctxa = (float*)d_ws;                       // [B,128] f32
    float*    ctxf = ctxa + (size_t)B * D_H;             // [B,128] f32
    _Float16* pwa  = (_Float16*)(ctxf + (size_t)B * D_H);
    _Float16* pwc  = pwa + PWA_ELEMS;

    pack_weights_kernel<<<(PWA_ELEMS + PWC_ELEMS + 255) / 256, 256, 0, stream>>>(
        w_a1, w_f1, pwa, pwc);
    ctx_proj_kernel<<<B / 16, 256, 0, stream>>>(ctx, b_a1, b_f1, pwc, ctxa, ctxf);
    news_main_kernel<<<B, 256, 0, stream>>>(kgat, idx_ent, idx_title, idx_type,
                                            tab_ent, tab_title, tab_type,
                                            pwa, w_a2, b_a2, w_f1, w_f2, b_f2,
                                            ctxa, ctxf, out_news, out_soft);
}
